// MultiHeadAttentionLayer_86251533238810
// MI455X (gfx1250) — matrix operand order repaired and checked by reading, not compiled, not run on hardware
//
#include <hip/hip_runtime.h>

#define BB 32
#define NN 151
#define NP 160             // N padded to 10 tiles of 16
#define DIMC 512
#define HH 8
#define DD 64
#define MROWS (BB * NN)    // 4832 (divisible by 32)
#define QKVC (3 * HH * DD) // 1536

typedef __attribute__((ext_vector_type(16))) __bf16 v16bf;
typedef __attribute__((ext_vector_type(8)))  float  v8f;
typedef __attribute__((ext_vector_type(4)))  int    v4i;

#if __has_builtin(__builtin_amdgcn_global_load_async_to_lds_b128) && \
    __has_builtin(__builtin_amdgcn_s_wait_asynccnt)
#define HAVE_ASYNC_LDS 1
#else
#define HAVE_ASYNC_LDS 0
#endif

#define GAS __attribute__((address_space(1)))
#define LAS __attribute__((address_space(3)))

__device__ __forceinline__ unsigned short f2bf(float f) {
  unsigned int u = __float_as_uint(f);
  u += 0x7FFFu + ((u >> 16) & 1u);   // round-to-nearest-even
  return (unsigned short)(u >> 16);
}

// A-matrix 16x32 bf16 fragment from row-major source (lane&15 = row).
// VGPR j<4: K = kh + 2j,2j+1 ; j>=4: K = 16 + kh + 2(j-4),+1 ; kh = (lane>>4)*8
__device__ __forceinline__ v16bf load_a_frag(const unsigned short* base, int ld,
                                             int row0, int k0) {
  const int lane = threadIdx.x & 31;
  const int r  = lane & 15;
  const int kh = (lane >> 4) << 3;
  const unsigned short* p = base + (size_t)(row0 + r) * ld + k0 + kh;
  union { v16bf v; unsigned short u[16]; } U;
#pragma unroll
  for (int j = 0; j < 4; ++j) {
    U.u[2 * j]     = p[2 * j];
    U.u[2 * j + 1] = p[2 * j + 1];
    U.u[2 * j + 8] = p[16 + 2 * j];
    U.u[2 * j + 9] = p[16 + 2 * j + 1];
  }
  return U.v;
}

// B-matrix 32x16 bf16 fragment: source is B-transposed row-major (lane&15 = column n).
// element s<8: K = kh + s ; s>=8: K = 16 + kh + (s-8) ; kh = (lane>>4)*8
__device__ __forceinline__ v16bf load_b_frag(const unsigned short* base, int ld,
                                             int row0, int k0) {
  const int lane = threadIdx.x & 31;
  const int n  = lane & 15;
  const int kh = (lane >> 4) << 3;
  const unsigned short* p = base + (size_t)(row0 + n) * ld + k0 + kh;
  union { v16bf v; unsigned short u[16]; } U;
#pragma unroll
  for (int t = 0; t < 16; ++t) U.u[t] = p[t < 8 ? t : t + 8];
  return U.v;
}

__device__ __forceinline__ v8f wmma_bf16(v16bf a, v16bf b, v8f c) {
  return __builtin_amdgcn_wmma_f32_16x16x32_bf16(false, a, false, b,
                                                 (short)0, c, false, false);
}

// ---------------- prep kernels ----------------

__global__ void pe_kernel(const float* __restrict__ angle_enc,
                          const float* __restrict__ cls_dist,
                          const float* __restrict__ cls_angle,
                          float* __restrict__ peD, float* __restrict__ peA) {
  const int d  = threadIdx.x;           // 0..63
  const int bn = blockIdx.x;            // b*NN + n
  const int b  = bn / NN, n = bn % NN;
  const float a  = (n == 0) ? 0.f : angle_enc[b * (NN - 1) + n - 1];
  const float ed = (n == 0) ? cls_dist[0]  : a;
  const float ea = (n == 0) ? cls_angle[0] : a;
  const float freq = __powf(10000.f, (float)(2 * (d >> 1)) * (1.f / 64.f));
  const float pd = floorf(ed * (float)NN / freq);
  const float pa = floorf(ea * (float)NN / freq);
  const float vd = ((d & 1) == 0) ? __sinf(pd) : __cosf(pd);
  const float va = ((d & 1) == 0) ? __sinf(pa) : __cosf(pa);
  const size_t idx = (size_t)bn * DD + d;
  peD[idx] = vd;
  peA[idx] = va;
}

__global__ void cvt_kernel(const float* __restrict__ src,
                           unsigned short* __restrict__ dst, int count) {
  int i = blockIdx.x * blockDim.x + threadIdx.x;
  if (i < count) dst[i] = f2bf(src[i]);
}

__global__ void cvtT_kernel(const float* __restrict__ src,
                            unsigned short* __restrict__ dst, int rows, int cols) {
  // src: rows x cols row-major -> dst: cols x rows row-major (transpose)
  int i = blockIdx.x * blockDim.x + threadIdx.x;
  if (i < rows * cols) {
    int rr = i / cols, c = i - rr * cols;
    dst[(size_t)c * rows + rr] = f2bf(src[i]);
  }
}

__global__ void zero_kernel(unsigned short* __restrict__ dst, int count) {
  int i = blockIdx.x * blockDim.x + threadIdx.x;
  if (i < count) dst[i] = 0;
}

// ---------------- QKV GEMM: (4832x512)x(512x1536), 32x64 tile per wave ----------

__global__ void __launch_bounds__(128)
qkv_gemm_kernel(const unsigned short* __restrict__ Xb,
                const unsigned short* __restrict__ WqkvT,
                const float* __restrict__ peD, const float* __restrict__ peA,
                unsigned short* __restrict__ Qb, unsigned short* __restrict__ Kb,
                unsigned short* __restrict__ Vt) {
  const int wave = threadIdx.x >> 5;
  const int m0 = blockIdx.y * 32;
  const int n0 = blockIdx.x * 256 + wave * 64;
  const v8f z = {0, 0, 0, 0, 0, 0, 0, 0};
  v8f acc[2][4];
#pragma unroll
  for (int mi = 0; mi < 2; ++mi)
#pragma unroll
    for (int t = 0; t < 4; ++t) acc[mi][t] = z;

#pragma unroll 2
  for (int kc = 0; kc < DIMC; kc += 32) {
    if (kc + 32 < DIMC) {
      __builtin_prefetch(Xb + (size_t)m0 * DIMC + kc + 32, 0, 1);
      __builtin_prefetch(WqkvT + (size_t)n0 * DIMC + kc + 32, 0, 1);
    }
    v16bf a0 = load_a_frag(Xb, DIMC, m0, kc);
    v16bf a1 = load_a_frag(Xb, DIMC, m0 + 16, kc);
    v16bf b0 = load_b_frag(WqkvT, DIMC, n0, kc);
    v16bf b1 = load_b_frag(WqkvT, DIMC, n0 + 16, kc);
    v16bf b2 = load_b_frag(WqkvT, DIMC, n0 + 32, kc);
    v16bf b3 = load_b_frag(WqkvT, DIMC, n0 + 48, kc);
    acc[0][0] = wmma_bf16(a0, b0, acc[0][0]);
    acc[0][1] = wmma_bf16(a0, b1, acc[0][1]);
    acc[0][2] = wmma_bf16(a0, b2, acc[0][2]);
    acc[0][3] = wmma_bf16(a0, b3, acc[0][3]);
    acc[1][0] = wmma_bf16(a1, b0, acc[1][0]);
    acc[1][1] = wmma_bf16(a1, b1, acc[1][1]);
    acc[1][2] = wmma_bf16(a1, b2, acc[1][2]);
    acc[1][3] = wmma_bf16(a1, b3, acc[1][3]);
  }

  const int lane = threadIdx.x & 31;
  const int ncol = lane & 15;
  const int mh   = (lane >> 4) << 3;
#pragma unroll
  for (int mi = 0; mi < 2; ++mi) {
#pragma unroll
    for (int t = 0; t < 4; ++t) {
      const int col = n0 + 16 * t + ncol;
      const int which = col >> 9;      // 0:q 1:k 2:v
      const int rem = col & 511;
      const int h = rem >> 6, d = rem & 63;
#pragma unroll
      for (int r = 0; r < 8; ++r) {
        const int m = m0 + mi * 16 + mh + r;
        const int b_ = m / NN, n = m - b_ * NN;
        const float v = acc[mi][t][r];
        const size_t pidx = ((size_t)b_ * NN + n) * DD + d;
        const size_t qk = (((size_t)(b_ * HH + h)) * NP + n) * DD + d;
        if (which == 0)      Qb[qk] = f2bf(v);
        else if (which == 1) Kb[qk] = f2bf(v + peD[pidx]);       // K' = K + Pd
        else Vt[(((size_t)(b_ * HH + h)) * DD + d) * NP + n] =
                 f2bf(v + peA[pidx]);                            // V'^T
      }
    }
  }
}

// ---------------- fused attention: one workgroup per (b,h) ----------------

__global__ void __launch_bounds__(320)
attn_kernel(const unsigned short* __restrict__ Qb,
            const unsigned short* __restrict__ Kb,
            const unsigned short* __restrict__ Vt,
            const float* __restrict__ peA,
            unsigned short* __restrict__ Ob) {
  // Single 51.2 KB buffer, used twice:
  //   phase 1: first NP*DD ushorts hold staged K' (row-major NPxDD)
  //   phase 2: whole buffer holds normalized probs P (row-major NPxNP, bf16)
  __shared__ unsigned short smem[NP * NP];
  const int bh = blockIdx.x;
  const int b = bh / HH, h = bh % HH;
  const int wave = threadIdx.x >> 5;
  const int lane = threadIdx.x & 31;
  const int tid  = threadIdx.x;
  const int m0 = wave * 16;                // this wave's 16 query rows
  const unsigned short* Qbh = Qb + (size_t)bh * NP * DD;
  const unsigned short* Kbh = Kb + (size_t)bh * NP * DD;
  const unsigned short* Vth = Vt + (size_t)bh * DD * NP;

  // ---- stage K' into LDS (20480 B = 1280 x 16B), async if available ----
  {
    v4i* gk = (v4i*)Kbh;                   // non-const, as the builtin requires
    v4i* lk = (v4i*)smem;
#if HAVE_ASYNC_LDS
#pragma unroll
    for (int i = 0; i < 4; ++i) {
      const int c = tid + i * 320;
      __builtin_amdgcn_global_load_async_to_lds_b128(
          (GAS v4i*)(gk + c), (LAS v4i*)(lk + c), 0, 0);
    }
    __builtin_amdgcn_s_wait_asynccnt(0);
#else
#pragma unroll
    for (int i = 0; i < 4; ++i) {
      const int c = tid + i * 320;
      lk[c] = gk[c];
    }
#endif
  }
  __syncthreads();
  const unsigned short* sK = smem;

  const v16bf aq0 = load_a_frag(Qbh, DD, m0, 0);
  const v16bf aq1 = load_a_frag(Qbh, DD, m0, 32);

  float Sf[10][8];
#pragma unroll
  for (int j = 0; j < 10; ++j) {
    v16bf bk0 = load_b_frag(sK, DD, 16 * j, 0);     // ds loads
    v16bf bk1 = load_b_frag(sK, DD, 16 * j, 32);
    v8f c = {0, 0, 0, 0, 0, 0, 0, 0};
    c = wmma_bf16(aq0, bk0, c);
    c = wmma_bf16(aq1, bk1, c);
#pragma unroll
    for (int r = 0; r < 8; ++r) Sf[j][r] = c[r];
  }

  const int ncol = lane & 15;
  const int mh   = (lane >> 4) << 3;
  const float scale = 0.125f;              // D^-1/2

  float mx[8], sm[8];
#pragma unroll
  for (int r = 0; r < 8; ++r) mx[r] = -3.0e38f;
#pragma unroll
  for (int j = 0; j < 10; ++j) {
    const int colg = 16 * j + ncol;
#pragma unroll
    for (int r = 0; r < 8; ++r) {
      float s = Sf[j][r] * scale;
      if (colg >= NN) s = -1.0e30f;        // mask pad keys
      Sf[j][r] = s;
      mx[r] = fmaxf(mx[r], s);
    }
  }
  // a row's 16 columns live within one 16-lane half -> reduce with width 16
#pragma unroll
  for (int r = 0; r < 8; ++r) {
#pragma unroll
    for (int off = 1; off < 16; off <<= 1)
      mx[r] = fmaxf(mx[r], __shfl_xor(mx[r], off, 16));
    sm[r] = 0.f;
  }
#pragma unroll
  for (int j = 0; j < 10; ++j) {
#pragma unroll
    for (int r = 0; r < 8; ++r) {
      float p = __expf(Sf[j][r] - mx[r]);
      Sf[j][r] = p;
      sm[r] += p;
    }
  }
#pragma unroll
  for (int r = 0; r < 8; ++r) {
#pragma unroll
    for (int off = 1; off < 16; off <<= 1)
      sm[r] += __shfl_xor(sm[r], off, 16);
    sm[r] = 1.f / sm[r];
  }

  __syncthreads();   // K' staging region is dead; safe to overwrite with P

#pragma unroll
  for (int j = 0; j < 10; ++j) {
#pragma unroll
    for (int r = 0; r < 8; ++r)
      smem[(size_t)(m0 + mh + r) * NP + 16 * j + ncol] = f2bf(Sf[j][r] * sm[r]);
  }
  __syncthreads();

  // O = P x V'  (K dim = 160, 5 chunks of 32; 4 d-tiles of 16)
#pragma unroll
  for (int dt = 0; dt < 4; ++dt) {
    v8f o = {0, 0, 0, 0, 0, 0, 0, 0};
#pragma unroll
    for (int c = 0; c < 5; ++c) {
      v16bf ap = load_a_frag(smem, NP, m0, 32 * c);  // ds loads
      v16bf bv = load_b_frag(Vth, NP, 16 * dt, 32 * c);
      o = wmma_bf16(ap, bv, o);
    }
#pragma unroll
    for (int r = 0; r < 8; ++r) {
      const int n = m0 + mh + r;
      if (n < NN) {
        const int d = 16 * dt + ncol;
        const float val = o[r] - peA[((size_t)b * NN + n) * DD + d]; // -Pa[n]
        Ob[((size_t)b * NN + n) * (HH * DD) + h * DD + d] = f2bf(val);
      }
    }
  }
}

// ---------------- output GEMM: (4832x512)x(512x512) + bias, 32x64/wave -------

__global__ void __launch_bounds__(128)
out_gemm_kernel(const unsigned short* __restrict__ Ob,
                const unsigned short* __restrict__ WoT,
                const float* __restrict__ b_out,
                float* __restrict__ out) {
  const int wave = threadIdx.x >> 5;
  const int m0 = blockIdx.y * 32;
  const int n0 = blockIdx.x * 256 + wave * 64;
  const v8f z = {0, 0, 0, 0, 0, 0, 0, 0};
  v8f acc[2][4];
#pragma unroll
  for (int mi = 0; mi < 2; ++mi)
#pragma unroll
    for (int t = 0; t < 4; ++t) acc[mi][t] = z;

#pragma unroll 2
  for (int kc = 0; kc < DIMC; kc += 32) {
    if (kc + 32 < DIMC) {
      __builtin_prefetch(Ob + (size_t)m0 * DIMC + kc + 32, 0, 1);
      __builtin_prefetch(WoT + (size_t)n0 * DIMC + kc + 32, 0, 1);
    }
    v16bf a0 = load_a_frag(Ob, DIMC, m0, kc);
    v16bf a1 = load_a_frag(Ob, DIMC, m0 + 16, kc);
    v16bf b0 = load_b_frag(WoT, DIMC, n0, kc);
    v16bf b1 = load_b_frag(WoT, DIMC, n0 + 16, kc);
    v16bf b2 = load_b_frag(WoT, DIMC, n0 + 32, kc);
    v16bf b3 = load_b_frag(WoT, DIMC, n0 + 48, kc);
    acc[0][0] = wmma_bf16(a0, b0, acc[0][0]);
    acc[0][1] = wmma_bf16(a0, b1, acc[0][1]);
    acc[0][2] = wmma_bf16(a0, b2, acc[0][2]);
    acc[0][3] = wmma_bf16(a0, b3, acc[0][3]);
    acc[1][0] = wmma_bf16(a1, b0, acc[1][0]);
    acc[1][1] = wmma_bf16(a1, b1, acc[1][1]);
    acc[1][2] = wmma_bf16(a1, b2, acc[1][2]);
    acc[1][3] = wmma_bf16(a1, b3, acc[1][3]);
  }

  const int lane = threadIdx.x & 31;
  const int ncol = lane & 15;
  const int mh   = (lane >> 4) << 3;
#pragma unroll
  for (int mi = 0; mi < 2; ++mi) {
#pragma unroll
    for (int t = 0; t < 4; ++t) {
      const int col = n0 + 16 * t + ncol;
      const float bias = b_out[col];
#pragma unroll
      for (int r = 0; r < 8; ++r) {
        const int m = m0 + mi * 16 + mh + r;
        out[(size_t)m * DIMC + col] = acc[mi][t][r] + bias;
      }
    }
  }
}

// ---------------- launch ----------------

extern "C" void kernel_launch(void* const* d_in, const int* in_sizes, int n_in,
                              void* d_out, int out_size, void* d_ws, size_t ws_size,
                              hipStream_t stream) {
  (void)in_sizes; (void)n_in; (void)out_size; (void)ws_size;
  const float* x         = (const float*)d_in[0];
  const float* angle_enc = (const float*)d_in[1];
  const float* W_qkv     = (const float*)d_in[2];
  const float* W_out     = (const float*)d_in[3];
  const float* b_out     = (const float*)d_in[4];
  const float* cls_dist  = (const float*)d_in[5];
  const float* cls_angle = (const float*)d_in[6];
  float* out = (float*)d_out;

  char* ws = (char*)d_ws;
  size_t off = 0;
  auto carve = [&](size_t bytes) -> char* {
    char* p = ws + off;
    off += (bytes + 255) & ~(size_t)255;
    return p;
  };
  float* peD = (float*)carve((size_t)BB * NN * DD * 4);
  float* peA = (float*)carve((size_t)BB * NN * DD * 4);
  unsigned short* Xb    = (unsigned short*)carve((size_t)MROWS * DIMC * 2);
  unsigned short* WqkvT = (unsigned short*)carve((size_t)QKVC * DIMC * 2);
  unsigned short* WoT   = (unsigned short*)carve((size_t)DIMC * DIMC * 2);
  unsigned short* Qb    = (unsigned short*)carve((size_t)BB * HH * NP * DD * 2);
  unsigned short* Kb    = (unsigned short*)carve((size_t)BB * HH * NP * DD * 2);
  unsigned short* Vt    = (unsigned short*)carve((size_t)BB * HH * NP * DD * 2);
  unsigned short* Ob    = (unsigned short*)carve((size_t)MROWS * DIMC * 2);

  {
    const int cnt = BB * HH * NP * DD;
    const int blk = (cnt + 255) / 256;
    zero_kernel<<<blk, 256, 0, stream>>>(Qb, cnt);
    zero_kernel<<<blk, 256, 0, stream>>>(Kb, cnt);
    zero_kernel<<<blk, 256, 0, stream>>>(Vt, cnt);
  }
  pe_kernel<<<BB * NN, DD, 0, stream>>>(angle_enc, cls_dist, cls_angle, peD, peA);
  {
    const int cnt = MROWS * DIMC;
    cvt_kernel<<<(cnt + 255) / 256, 256, 0, stream>>>(x, Xb, cnt);
  }
  {
    const int cnt = DIMC * QKVC;
    cvtT_kernel<<<(cnt + 255) / 256, 256, 0, stream>>>(W_qkv, WqkvT, DIMC, QKVC);
  }
  {
    const int cnt = DIMC * DIMC;
    cvtT_kernel<<<(cnt + 255) / 256, 256, 0, stream>>>(W_out, WoT, DIMC, DIMC);
  }
  qkv_gemm_kernel<<<dim3(QKVC / 256, MROWS / 32), 128, 0, stream>>>(
      Xb, WqkvT, peD, peA, Qb, Kb, Vt);
  attn_kernel<<<BB * HH, 320, 0, stream>>>(Qb, Kb, Vt, peA, Ob);
  out_gemm_kernel<<<dim3(DIMC / 256, MROWS / 32), 128, 0, stream>>>(Ob, WoT, b_out, out);
}
